// DRTScorer_84447646974678
// MI455X (gfx1250) — matrix operand order repaired and checked by reading, not compiled, not run on hardware
//
#include <hip/hip_runtime.h>

// ---------------------------------------------------------------------------
// MI455X (gfx1250) implementation.
// Pipeline:
//   K0  pack_w_kernel      : W1, W2 (f32) -> bf16 WMMA B-fragment blobs in ws
//   K1  query_encode_kernel: 64 queries, full f32 MLP + alphas/softmax; writes
//                            alpha-folded normalized query codes as bf16 WMMA
//                            A-fragment blob (QW, 64x384)
//   K2  doc_encode_kernel  : per 16-doc tile: TDM global->LDS stage, bf16 WMMA
//                            GEMM1 (relu) + GEMM2, f32 slot-normalize, pack to
//                            bf16 WMMA B-fragment blob (D^T codes)
//   K3  score_kernel       : score = QW(64x384) @ Dcodes^T via WMMA, f32 out
// ---------------------------------------------------------------------------

typedef __attribute__((ext_vector_type(16))) __bf16        v16bf;
typedef __attribute__((ext_vector_type(8)))  float         v8f;
typedef __attribute__((ext_vector_type(4)))  unsigned int  u32x4;
typedef __attribute__((ext_vector_type(4)))  int           i32x4;
typedef __attribute__((ext_vector_type(8)))  int           i32x8;

union FragAB {
  v16bf bf;
  u32x4 q[2];     // 32 bytes per lane = 8 VGPRs
};

__device__ __forceinline__ v8f wmma_bf16(v16bf a, v16bf b, v8f c) {
  // (neg_a, A, neg_b, B, c_mod, C, reuse_a, reuse_b)
  return __builtin_amdgcn_wmma_f32_16x16x32_bf16(false, a, false, b,
                                                 (short)0, c, false, false);
}

__device__ __forceinline__ unsigned short bf16_bits(float f) {
  union { __bf16 h; unsigned short u; } cvt;
  cvt.h = (__bf16)f;   // RNE
  return cvt.u;
}

// ---------------------------------------------------------------------------
// K0: pack a (Kdim x Ndim) row-major f32 weight matrix into bf16 WMMA
// B-fragment blobs. Fragment f = n0t*ktiles + kt is 1024B: for lane l, dword v
// it holds elements (k = kt*32 + (l/16)*8 + 16*(v/4) + 2*(v%4) (+1), n = n0t*16 + l%16).
// One thread produces one packed dword. Grid: Kdim*Ndim/2/256 blocks x 256.
// ---------------------------------------------------------------------------
__global__ __launch_bounds__(256) void pack_w_kernel(
    const float* __restrict__ src, unsigned int* __restrict__ dst,
    int Ndim, int ktiles) {
  int gid  = blockIdx.x * 256 + threadIdx.x;
  int f    = gid >> 8;        // fragment index (256 dwords each)
  int rem  = gid & 255;
  int lane = rem >> 3, v = rem & 7;
  int n0t  = f / ktiles, kt = f % ktiles;
  int n    = n0t * 16 + (lane & 15);
  int k0   = kt * 32 + ((lane >> 4) << 3) + ((v >> 2) << 4) + ((v & 3) << 1);
  float f0 = src[(size_t)k0 * Ndim + n];
  float f1 = src[(size_t)(k0 + 1) * Ndim + n];
  dst[gid] = (unsigned)bf16_bits(f0) | ((unsigned)bf16_bits(f1) << 16);
}

// ---------------------------------------------------------------------------
// K1: query encoder (tiny, exact f32). One block per query, 512 threads.
// Writes alpha-folded normalized codes into the WMMA A-fragment blob:
// for query b (m = b%16, mt = b/16) and dim t: frag = mt*12 + t/32,
// lane = m + 16*((t%16)/8), v = 4*((t%32)/16) + ((t%8)/2), slot = t%2.
// ---------------------------------------------------------------------------
__global__ __launch_bounds__(512) void query_encode_kernel(
    const float* __restrict__ q,   const float* __restrict__ W1,
    const float* __restrict__ b1,  const float* __restrict__ W2,
    const float* __restrict__ b2,  const float* __restrict__ Wa1,
    const float* __restrict__ ba1, const float* __restrict__ Wa2,
    const float* __restrict__ ba2, char* __restrict__ qwblob) {
  __shared__ float xs[384], hs[512], ah[64], sv[384], lg[6], al[6], ns[6];
  const int t = threadIdx.x, b = blockIdx.x;
  if (t < 384) xs[t] = q[b * 384 + t];
  __syncthreads();
  {  // h = relu(x @ W1 + b1), all 512 threads
    float a = b1[t];
    for (int k = 0; k < 384; ++k) a += xs[k] * W1[k * 512 + t];
    hs[t] = fmaxf(a, 0.f);
  }
  if (t < 64) {  // attention hidden
    float a = ba1[t];
    for (int k = 0; k < 384; ++k) a += xs[k] * Wa1[k * 64 + t];
    ah[t] = fmaxf(a, 0.f);
  }
  __syncthreads();
  if (t < 384) {  // s = h @ W2 + b2
    float s = b2[t];
    for (int k = 0; k < 512; ++k) s += hs[k] * W2[k * 384 + t];
    sv[t] = s;
  }
  if (t >= 504 && t < 510) {  // attention logits (6)
    int kk = t - 504;
    float g = ba2[kk];
    for (int i = 0; i < 64; ++i) g += ah[i] * Wa2[i * 6 + kk];
    lg[kk] = g;
  }
  __syncthreads();
  if (t < 6) {  // per-slot inverse norms
    float s2 = 0.f;
    for (int i = 0; i < 64; ++i) { float x = sv[t * 64 + i]; s2 += x * x; }
    ns[t] = rsqrtf(s2 + 1e-12f);
  }
  if (t == 0) {  // softmax over 6 slots
    float m = lg[0];
    for (int k = 1; k < 6; ++k) m = fmaxf(m, lg[k]);
    float e[6], s = 0.f;
    for (int k = 0; k < 6; ++k) { e[k] = expf(lg[k] - m); s += e[k]; }
    for (int k = 0; k < 6; ++k) al[k] = e[k] / s;
  }
  __syncthreads();
  if (t < 384) {  // qw = alpha * normalized sub-vector, packed A-fragment
    int slot = t >> 6;
    float qw = sv[t] * ns[slot] * al[slot];
    int mt = b >> 4, mr = b & 15;
    int kt = t >> 5, kk = t & 31;
    int laneA = mr + 16 * ((kk >> 3) & 1);
    int v = 4 * (kk >> 4) + ((kk & 7) >> 1);
    *(unsigned short*)(qwblob + (((size_t)(mt * 12 + kt)) << 10) +
                       laneA * 32 + v * 4 + (kk & 1) * 2) = bf16_bits(qw);
  }
}

// ---------------------------------------------------------------------------
// K2: doc encoder. 16 docs per workgroup, 256 threads = 8 waves.
// LDS layout (53632 B):
//   [0,24576)      Xraw (16x384 f32, TDM dest)   -> reused as Sbuf (f32 S)
//   [24576,36864)  Xa A-fragments (12 x 1KB)     -> reused as Dfrag pack
//   [36864,53248)  Ha A-fragments (16 x 1KB)
//   [53248,53632)  inverse norms (16 docs x 6 slots f32)
// ---------------------------------------------------------------------------
#define XRAW_OFF  0
#define SBUF_OFF  0
#define XA_OFF    24576
#define DFRAG_OFF 24576
#define HA_OFF    36864
#define NORM_OFF  53248
#define K2_LDS    53632

__global__ __launch_bounds__(256) void doc_encode_kernel(
    const float* __restrict__ doc, const char* __restrict__ w1blob,
    const char* __restrict__ w2blob, const float* __restrict__ b1,
    const float* __restrict__ b2, char* __restrict__ docblob, int P) {
  __shared__ __align__(16) char sm[K2_LDS];
  const int tid = threadIdx.x;
  const int lane = tid & 31, wave = tid >> 5;
  const int p0 = blockIdx.x * 16;

  // --- Stage X tile (16 x 384 f32) global -> LDS via Tensor Data Mover -----
  if (wave == 0) {
    unsigned long long gaddr =
        (unsigned long long)(const void*)(doc + (size_t)p0 * 384);
    unsigned int ldsoff = (unsigned int)(unsigned long long)(&sm[XRAW_OFF]);
    unsigned int dim1 = (unsigned int)(P - p0);  // remaining rows (OOB -> 0)
    u32x4 g0; i32x8 g1; i32x4 g2 = {0, 0, 0, 0}, g3 = {0, 0, 0, 0};
    i32x8 g4 = {0, 0, 0, 0, 0, 0, 0, 0};
    g0[0] = 1u;                                        // count=1, no gather
    g0[1] = ldsoff;                                    // lds_addr
    g0[2] = (unsigned int)(gaddr & 0xFFFFFFFFull);     // global_addr lo
    g0[3] = (unsigned int)((gaddr >> 32) & 0x1FFFFFFull) | 0x80000000u; // type=2
    g1[0] = 0x00020000;                 // wg_mask=0, data_size=2 (4 bytes)
    g1[1] = (int)(384u << 16);          // tensor_dim0[15:0] = 384
    g1[2] = (int)((dim1 & 0xFFFFu) << 16);            // tensor_dim1 lo16
    g1[3] = (int)((dim1 >> 16) | (384u << 16));       // dim1 hi | tile_dim0=384
    g1[4] = 16;                         // tile_dim1 = 16 rows
    g1[5] = 384;                        // tensor_dim0_stride = 384
    g1[6] = 0;
    g1[7] = 0;
    __builtin_amdgcn_tensor_load_to_lds(g0, g1, g2, g3, g4, 0);
    __builtin_amdgcn_s_wait_tensorcnt(0);
  }
  __syncthreads();

  // --- Convert Xraw -> bf16 A-fragments in LDS -----------------------------
  for (int i = 0; i < 24; ++i) {
    int idx = tid + 256 * i;                 // 16*384 = 6144 elements
    int d = idx / 384, k = idx % 384;
    float x = ((const float*)(sm + XRAW_OFF))[idx];
    int kt = k >> 5, kk = k & 31;
    int laneA = d + 16 * ((kk >> 3) & 1);
    int v = 4 * (kk >> 4) + ((kk & 7) >> 1);
    *(unsigned short*)(sm + XA_OFF + (kt << 10) + laneA * 32 + v * 4 +
                       (kk & 1) * 2) = bf16_bits(x);
  }
  __syncthreads();

  // --- GEMM1: H(16x512) = relu(X @ W1 + b1), write Ha fragments ------------
  for (int j = 0; j < 4; ++j) {
    int nt = wave + 8 * j;                   // 32 n-tiles of 16
    float bn = b1[nt * 16 + (lane & 15)];
    v8f acc = {bn, bn, bn, bn, bn, bn, bn, bn};
#pragma unroll
    for (int kt = 0; kt < 12; ++kt) {
      FragAB a, bf;
      const u32x4* ap = (const u32x4*)(sm + XA_OFF + (kt << 10) + lane * 32);
      a.q[0] = ap[0]; a.q[1] = ap[1];
      const u32x4* bp =
          (const u32x4*)(w1blob + (((size_t)(nt * 12 + kt)) << 10) + lane * 32);
      bf.q[0] = bp[0]; bf.q[1] = bp[1];
      acc = wmma_bf16(a.bf, bf.bf, acc);
    }
    int n = lane & 15;
    int kt2 = nt >> 1;
    int v2 = 4 * (nt & 1) + ((n & 7) >> 1);
    int laneBase = 8 * (lane >> 4) + 16 * ((n >> 3) & 1);
#pragma unroll
    for (int r = 0; r < 8; ++r) {
      float h = fmaxf(acc[r], 0.f);
      *(unsigned short*)(sm + HA_OFF + (kt2 << 10) + (laneBase + r) * 32 +
                         v2 * 4 + (n & 1) * 2) = bf16_bits(h);
    }
  }
  __syncthreads();

  // --- GEMM2: S(16x384) = H @ W2 + b2, f32 into Sbuf -----------------------
  for (int j = 0; j < 3; ++j) {
    int nt = wave + 8 * j;                   // 24 n-tiles of 16
    float bn = b2[nt * 16 + (lane & 15)];
    v8f acc = {bn, bn, bn, bn, bn, bn, bn, bn};
#pragma unroll
    for (int kt = 0; kt < 16; ++kt) {
      FragAB a, bf;
      const u32x4* ap = (const u32x4*)(sm + HA_OFF + (kt << 10) + lane * 32);
      a.q[0] = ap[0]; a.q[1] = ap[1];
      const u32x4* bp =
          (const u32x4*)(w2blob + (((size_t)(nt * 16 + kt)) << 10) + lane * 32);
      bf.q[0] = bp[0]; bf.q[1] = bp[1];
      acc = wmma_bf16(a.bf, bf.bf, acc);
    }
    float* sb = (float*)(sm + SBUF_OFF);
    int n2 = nt * 16 + (lane & 15);
#pragma unroll
    for (int r = 0; r < 8; ++r) sb[(r + 8 * (lane >> 4)) * 384 + n2] = acc[r];
  }
  __syncthreads();

  // --- Per-slot inverse norms (16 docs x 6 slots) --------------------------
  if (tid < 96) {
    int d = tid / 6, s = tid % 6;
    const float* sb = (const float*)(sm + SBUF_OFF);
    float s2 = 0.f;
    for (int i = 0; i < 64; ++i) { float x = sb[d * 384 + s * 64 + i]; s2 += x * x; }
    ((float*)(sm + NORM_OFF))[tid] = rsqrtf(s2 + 1e-12f);
  }
  __syncthreads();

  // --- Normalize + pack to bf16 B-fragments in LDS -------------------------
  for (int i = 0; i < 24; ++i) {
    int idx = tid + 256 * i;
    int d = idx / 384, k = idx % 384;
    float val = ((const float*)(sm + SBUF_OFF))[idx] *
                ((const float*)(sm + NORM_OFF))[d * 6 + (k >> 6)];
    int fr = k >> 5, kk = k & 31;
    int laneB = d + 16 * ((kk >> 3) & 1);
    *(unsigned short*)(sm + DFRAG_OFF + (fr << 10) + laneB * 32 +
                       (4 * (kk >> 4) + ((kk & 7) >> 1)) * 4 + (kk & 1) * 2) = bf16_bits(val);
  }
  __syncthreads();

  // --- Stream packed fragments to global blob (coalesced b128) -------------
  {
    u32x4* dst = (u32x4*)(docblob + (size_t)blockIdx.x * 12288);
    const u32x4* srcp = (const u32x4*)(sm + DFRAG_OFF);
    for (int i = 0; i < 3; ++i) dst[tid + 256 * i] = srcp[tid + 256 * i];
  }
}

// ---------------------------------------------------------------------------
// K3: scoring GEMM. out(64 x P) = QW @ Dcodes^T. 8 waves: wave handles
// query m-tile (wave&3) over 8 doc n-tiles of chunk (wave>>2). All 12 A
// fragments held in VGPRs (96 VGPRs), doc B fragments streamed from the blob.
// ---------------------------------------------------------------------------
__global__ __launch_bounds__(256) void score_kernel(
    const char* __restrict__ qwblob, const char* __restrict__ docblob,
    float* __restrict__ out, int P, int ntiles) {
  const int tid = threadIdx.x, lane = tid & 31, wave = tid >> 5;
  const int mt = wave & 3, c = wave >> 2;
  FragAB A[12];
#pragma unroll
  for (int kt = 0; kt < 12; ++kt) {
    const u32x4* p =
        (const u32x4*)(qwblob + (((size_t)(mt * 12 + kt)) << 10) + lane * 32);
    A[kt].q[0] = p[0]; A[kt].q[1] = p[1];
  }
  for (int j = 0; j < 8; ++j) {
    int tix = blockIdx.x * 16 + c * 8 + j;
    if (tix >= ntiles) break;
    v8f acc = {0.f, 0.f, 0.f, 0.f, 0.f, 0.f, 0.f, 0.f};
#pragma unroll
    for (int kt = 0; kt < 12; ++kt) {
      FragAB b;
      const u32x4* p = (const u32x4*)(docblob +
                       (((size_t)tix * 12 + kt) << 10) + lane * 32);
      b.q[0] = p[0]; b.q[1] = p[1];
      acc = wmma_bf16(A[kt].bf, b.bf, acc);
    }
    int docidx = tix * 16 + (lane & 15);
    int mbase = mt * 16 + 8 * (lane >> 4);
    if (docidx < P) {
#pragma unroll
      for (int r = 0; r < 8; ++r)
        out[(size_t)(mbase + r) * P + docidx] = acc[r];
    }
  }
}

// ---------------------------------------------------------------------------
extern "C" void kernel_launch(void* const* d_in, const int* in_sizes, int n_in,
                              void* d_out, int out_size, void* d_ws,
                              size_t ws_size, hipStream_t stream) {
  const float* q   = (const float*)d_in[0];
  const float* doc = (const float*)d_in[1];
  const float* W1  = (const float*)d_in[2];
  const float* b1  = (const float*)d_in[3];
  const float* W2  = (const float*)d_in[4];
  const float* b2  = (const float*)d_in[5];
  const float* Wa1 = (const float*)d_in[6];
  const float* ba1 = (const float*)d_in[7];
  const float* Wa2 = (const float*)d_in[8];
  const float* ba2 = (const float*)d_in[9];
  float* out = (float*)d_out;

  const int P = in_sizes[1] / 384;           // 100000
  const int B = in_sizes[0] / 384;           // 64 (4 m-tiles assumed)
  const int ntiles = (P + 15) / 16;

  char* ws      = (char*)d_ws;
  char* w1blob  = ws;                         // 384 KB
  char* w2blob  = ws + 393216;                // 384 KB
  char* qwblob  = ws + 786432;                // 48 KB
  char* docblob = ws + 835584;                // ntiles * 12 KB (~76.8 MB)

  pack_w_kernel<<<384, 256, 0, stream>>>(W1, (unsigned int*)w1blob, 512, 12);
  pack_w_kernel<<<384, 256, 0, stream>>>(W2, (unsigned int*)w2blob, 384, 16);
  query_encode_kernel<<<B, 512, 0, stream>>>(q, W1, b1, W2, b2, Wa1, ba1,
                                             Wa2, ba2, qwblob);
  doc_encode_kernel<<<ntiles, 256, 0, stream>>>(doc, w1blob, w2blob, b1, b2,
                                                docblob, P);
  score_kernel<<<(ntiles + 15) / 16, 256, 0, stream>>>(qwblob, docblob, out,
                                                       P, ntiles);
}
